// EGNNLayer_46076409151882
// MI455X (gfx1250) — matrix operand order repaired and checked by reading, not compiled, not run on hardware
//
#include <hip/hip_runtime.h>
#include <hip/hip_bf16.h>

#define ND 128
#define ED 16
#define HD 256
#define KE 288   // padded edge-feature K: 2*128+1+16 = 273 -> 288 (mult of 32)
#define KN 384   // node concat K: 128 + 256
#define EB 32    // edges per block (2 M-tiles)
#define NB 32    // nodes per block (2 M-tiles)

typedef __attribute__((ext_vector_type(16))) __bf16       v16bf;
typedef __attribute__((ext_vector_type(8)))  float        v8f;
typedef __attribute__((ext_vector_type(4)))  unsigned int v4u;

union Frag { v16bf bf; v4u u[2]; };

__device__ __forceinline__ unsigned short f2bf(float f) {
  union { float f; unsigned int u; } v; v.f = f;
  unsigned int r = v.u + 0x7FFFu + ((v.u >> 16) & 1u);   // RNE
  return (unsigned short)(r >> 16);
}
__device__ __forceinline__ float bf2f(unsigned short u) {
  union { unsigned int u; float f; } v; v.u = ((unsigned int)u) << 16; return v.f;
}
__device__ __forceinline__ float silu_f(float v) { return v / (1.f + __expf(-v)); }

// ---------------------------------------------------------------------------
// 2x2-tile WMMA GEMM: D[32, n0:n0+32] += A[32,KP](LDS bf16) * B[KP,NC](global)
// A frag (16-bit 16x32, ISA 7.12.2): lane m=l&15, K-half by l>=16, 8 VGPRs.
// B frag (16-bit 32x16): lane n=l&15, same K order as A (table order).
// Per k-step: 4 ds_load_b128 (A, 2 M-tiles) + 32 16-bit global loads per lane (B, 2 N-tiles)
// feeding 4 v_wmma -> 2x the WMMA density of the 1x2 version.
// ---------------------------------------------------------------------------
template <int KP>
__device__ __forceinline__ void wmma_gemm2x2(const unsigned short* __restrict__ Abuf,
                                             const unsigned short* __restrict__ B,
                                             int NC, int n0, int lane,
                                             v8f& a00, v8f& a01, v8f& a10, v8f& a11) {
  const int m    = lane & 15;
  const int koff = (lane >> 4) << 3;
#pragma unroll
  for (int k0 = 0; k0 < KP; k0 += 32) {
    Frag A0, A1, B0, B1;
    A0.u[0] = *(const v4u*)(Abuf + m * KP + k0 + koff);
    A0.u[1] = *(const v4u*)(Abuf + m * KP + k0 + 16 + koff);
    A1.u[0] = *(const v4u*)(Abuf + (m + 16) * KP + k0 + koff);
    A1.u[1] = *(const v4u*)(Abuf + (m + 16) * KP + k0 + 16 + koff);
#pragma unroll
    for (int i = 0; i < 8; ++i) {
      const int kk = k0 + koff + (i & 3) * 2 + (i >> 2) * 16;
      const unsigned short* brow = B + (size_t)kk * NC + n0 + m;
      B0.u[i >> 2][i & 3] = (unsigned)brow[0] | ((unsigned)brow[NC] << 16);
      B1.u[i >> 2][i & 3] = (unsigned)brow[16] | ((unsigned)brow[NC + 16] << 16);
    }
    a00 = __builtin_amdgcn_wmma_f32_16x16x32_bf16(false, A0.bf, false, B0.bf, (short)0, a00, false, false);
    a01 = __builtin_amdgcn_wmma_f32_16x16x32_bf16(false, A0.bf, false, B1.bf, (short)0, a01, false, false);
    a10 = __builtin_amdgcn_wmma_f32_16x16x32_bf16(false, A1.bf, false, B0.bf, (short)0, a10, false, false);
    a11 = __builtin_amdgcn_wmma_f32_16x16x32_bf16(false, A1.bf, false, B1.bf, (short)0, a11, false, false);
  }
}

// D tile (f32 16x16): lane n=l&15, row = mb0 + r + 8*(l>=16). Bias+SiLU -> LDS bf16.
__device__ __forceinline__ void store_silu_bf16(const v8f& acc, const float* __restrict__ bias,
                                                unsigned short* __restrict__ out /*[32][HD]*/,
                                                int mb0, int n0, int lane) {
  const int n  = n0 + (lane & 15);
  const int mb = mb0 + ((lane >> 4) << 3);
  const float bb = bias[n];
#pragma unroll
  for (int r = 0; r < 8; ++r) out[(mb + r) * HD + n] = f2bf(silu_f(acc[r] + bb));
}

// ---------------------------------------------------------------------------
// Edge kernel: 32 edges / block, 8 waves. Full edge-MLP pipeline + scatters.
// ---------------------------------------------------------------------------
__global__ __launch_bounds__(256)
void egnn_edge_kernel(const float* __restrict__ h, const float* __restrict__ x,
                      const float* __restrict__ edge_attr, const int* __restrict__ edge_idx,
                      const unsigned short* __restrict__ We1b, const float* __restrict__ be1,
                      const unsigned short* __restrict__ We2b, const float* __restrict__ be2,
                      const float* __restrict__ Wa, const float* __restrict__ ba,
                      const unsigned short* __restrict__ Wc1b, const float* __restrict__ bc1,
                      const float* __restrict__ Wc2,
                      float* __restrict__ agg, float* __restrict__ coord_agg, int Ne) {
  __shared__ unsigned short feat[EB * KE];
  __shared__ unsigned short buf1[EB * HD];
  __shared__ unsigned short buf2[EB * HD];
  __shared__ float sdiff[EB][3];
  __shared__ float sdist[EB];
  __shared__ float satt[EB];
  __shared__ float scw[EB];
  __shared__ int   srow[EB], scol[EB];

  const int tid  = threadIdx.x;
  const int lane = tid & 31;
  const int wv   = tid >> 5;
  const int e0   = blockIdx.x * EB;

  // warm L2 for the first (largest) weight matrix -> global_prefetch_b8
  __builtin_prefetch((const char*)We1b + (size_t)tid * 576, 0, 1);

  if (tid < EB) {
    int eg = e0 + tid;
    int r = 0, c = 0; float dx = 0, dy = 0, dz = 0;
    if (eg < Ne) {
      r = edge_idx[eg];
      c = edge_idx[Ne + eg];
      dx = x[r * 3 + 0] - x[c * 3 + 0];
      dy = x[r * 3 + 1] - x[c * 3 + 1];
      dz = x[r * 3 + 2] - x[c * 3 + 2];
    }
    srow[tid] = r; scol[tid] = c;
    sdiff[tid][0] = dx; sdiff[tid][1] = dy; sdiff[tid][2] = dz;
    sdist[tid] = dx * dx + dy * dy + dz * dz;
  }
  __syncthreads();

  // Gather [h[row] | h[col] | d2 | edge_attr | pad] as bf16
  for (int i = tid; i < EB * KE; i += 256) {
    int e = i / KE, c = i - e * KE;
    int eg = e0 + e;
    float v = 0.f;
    if (eg < Ne) {
      if (c < ND)                   v = h[(size_t)srow[e] * ND + c];
      else if (c < 2 * ND)          v = h[(size_t)scol[e] * ND + (c - ND)];
      else if (c == 2 * ND)         v = sdist[e];
      else if (c < 2 * ND + 1 + ED) v = edge_attr[(size_t)eg * ED + (c - (2 * ND + 1))];
    }
    feat[i] = f2bf(v);
  }
  __syncthreads();

  { // GEMM1: feat @ We1 -> silu -> buf1
    v8f a00 = {0,0,0,0,0,0,0,0}, a01 = a00, a10 = a00, a11 = a00;
    wmma_gemm2x2<KE>(feat, We1b, HD, wv * 32, lane, a00, a01, a10, a11);
    store_silu_bf16(a00, be1, buf1, 0,  wv * 32,      lane);
    store_silu_bf16(a01, be1, buf1, 0,  wv * 32 + 16, lane);
    store_silu_bf16(a10, be1, buf1, 16, wv * 32,      lane);
    store_silu_bf16(a11, be1, buf1, 16, wv * 32 + 16, lane);
  }
  __syncthreads();
  { // GEMM2: buf1 @ We2 -> silu -> msg in buf2
    v8f a00 = {0,0,0,0,0,0,0,0}, a01 = a00, a10 = a00, a11 = a00;
    wmma_gemm2x2<HD>(buf1, We2b, HD, wv * 32, lane, a00, a01, a10, a11);
    store_silu_bf16(a00, be2, buf2, 0,  wv * 32,      lane);
    store_silu_bf16(a01, be2, buf2, 0,  wv * 32 + 16, lane);
    store_silu_bf16(a10, be2, buf2, 16, wv * 32,      lane);
    store_silu_bf16(a11, be2, buf2, 16, wv * 32 + 16, lane);
  }
  __syncthreads();
  { // attention: sigmoid(msg @ Wa + ba); 8 lanes per edge, wave32 shfl_xor reduce
    int e = tid >> 3, j = tid & 7;
    float s = 0.f;
#pragma unroll
    for (int t = 0; t < 32; ++t) { int c = j + t * 8; s += bf2f(buf2[e * HD + c]) * Wa[c]; }
#pragma unroll
    for (int off = 4; off; off >>= 1) s += __shfl_xor(s, off, 32);
    if (j == 0) satt[e] = 1.f / (1.f + __expf(-(s + ba[0])));
  }
  __syncthreads();
  { // msg *= att
    int e = tid >> 3, j = tid & 7;
    float a = satt[e];
#pragma unroll
    for (int t = 0; t < 32; ++t) {
      int c = j + t * 8;
      buf2[e * HD + c] = f2bf(bf2f(buf2[e * HD + c]) * a);
    }
  }
  __syncthreads();
  { // GEMM3: msg @ Wc1 -> silu -> buf1
    v8f a00 = {0,0,0,0,0,0,0,0}, a01 = a00, a10 = a00, a11 = a00;
    wmma_gemm2x2<HD>(buf2, Wc1b, HD, wv * 32, lane, a00, a01, a10, a11);
    store_silu_bf16(a00, bc1, buf1, 0,  wv * 32,      lane);
    store_silu_bf16(a01, bc1, buf1, 0,  wv * 32 + 16, lane);
    store_silu_bf16(a10, bc1, buf1, 16, wv * 32,      lane);
    store_silu_bf16(a11, bc1, buf1, 16, wv * 32 + 16, lane);
  }
  __syncthreads();
  { // coord weight: tanh(t2 @ Wc2)
    int e = tid >> 3, j = tid & 7;
    float s = 0.f;
#pragma unroll
    for (int t = 0; t < 32; ++t) { int c = j + t * 8; s += bf2f(buf1[e * HD + c]) * Wc2[c]; }
#pragma unroll
    for (int off = 4; off; off >>= 1) s += __shfl_xor(s, off, 32);
    if (j == 0) scw[e] = tanhf(s);
  }
  __syncthreads();
  // equivariant coord scatter
  if (tid < 3 * EB) {
    int e = tid / 3, k = tid - e * 3;
    int eg = e0 + e;
    if (eg < Ne) {
      float inv = rsqrtf(sdist[e] + 1e-8f);
      atomicAdd(&coord_agg[(size_t)srow[e] * 3 + k], scw[e] * sdiff[e][k] * inv);
    }
  }
  // message scatter (segment_sum over row)
  {
    int e = tid >> 3, j = tid & 7;
    int eg = e0 + e;
    if (eg < Ne) {
      float* dst = agg + (size_t)srow[e] * HD;
#pragma unroll
      for (int t = 0; t < 32; ++t) {
        int c = j + t * 8;
        atomicAdd(&dst[c], bf2f(buf2[e * HD + c]));
      }
    }
  }
}

// ---------------------------------------------------------------------------
// Node kernel: [h | agg] @ Wn1 -> silu -> @ Wn2 -> +h -> LayerNorm
// ---------------------------------------------------------------------------
__global__ __launch_bounds__(256)
void egnn_node_kernel(const float* __restrict__ h, const float* __restrict__ agg,
                      const unsigned short* __restrict__ Wn1b, const float* __restrict__ bn1,
                      const unsigned short* __restrict__ Wn2b, const float* __restrict__ bn2,
                      const float* __restrict__ gamma, const float* __restrict__ beta,
                      float* __restrict__ h_out, int Nn) {
  __shared__ unsigned short cat[NB * KN];
  __shared__ unsigned short buf[NB * HD];
  __shared__ float hr[NB * ND];

  const int tid = threadIdx.x, lane = tid & 31, wv = tid >> 5;
  const int nd0 = blockIdx.x * NB;

  for (int i = tid; i < NB * KN; i += 256) {
    int e = i / KN, c = i - e * KN;
    int node = nd0 + e;
    float v = 0.f;
    if (node < Nn) v = (c < ND) ? h[(size_t)node * ND + c] : agg[(size_t)node * HD + (c - ND)];
    cat[i] = f2bf(v);
  }
  __syncthreads();
  { // GEMM1: cat @ Wn1 -> silu -> buf
    v8f a00 = {0,0,0,0,0,0,0,0}, a01 = a00, a10 = a00, a11 = a00;
    wmma_gemm2x2<KN>(cat, Wn1b, HD, wv * 32, lane, a00, a01, a10, a11);
    store_silu_bf16(a00, bn1, buf, 0,  wv * 32,      lane);
    store_silu_bf16(a01, bn1, buf, 0,  wv * 32 + 16, lane);
    store_silu_bf16(a10, bn1, buf, 16, wv * 32,      lane);
    store_silu_bf16(a11, bn1, buf, 16, wv * 32 + 16, lane);
  }
  __syncthreads();
  { // GEMM2: buf @ Wn2 (256->128); wave wv owns N-tile wv*16, both M-tiles
    v8f acc0 = {0,0,0,0,0,0,0,0}, acc1 = acc0;
    const int m = lane & 15, koff = (lane >> 4) << 3;
#pragma unroll
    for (int k0 = 0; k0 < HD; k0 += 32) {
      Frag A0, A1, B0;
      A0.u[0] = *(const v4u*)(buf + m * HD + k0 + koff);
      A0.u[1] = *(const v4u*)(buf + m * HD + k0 + 16 + koff);
      A1.u[0] = *(const v4u*)(buf + (m + 16) * HD + k0 + koff);
      A1.u[1] = *(const v4u*)(buf + (m + 16) * HD + k0 + 16 + koff);
#pragma unroll
      for (int i = 0; i < 8; ++i) {
        const int kk = k0 + koff + (i & 3) * 2 + (i >> 2) * 16;
        const unsigned short* brow = Wn2b + (size_t)kk * ND + wv * 16 + m;
        B0.u[i >> 2][i & 3] = (unsigned)brow[0] | ((unsigned)brow[ND] << 16);
      }
      acc0 = __builtin_amdgcn_wmma_f32_16x16x32_bf16(false, A0.bf, false, B0.bf, (short)0, acc0, false, false);
      acc1 = __builtin_amdgcn_wmma_f32_16x16x32_bf16(false, A1.bf, false, B0.bf, (short)0, acc1, false, false);
    }
    const int n  = wv * 16 + (lane & 15);
    const int mb = (lane >> 4) << 3;
    const float bb = bn2[n];
#pragma unroll
    for (int r = 0; r < 8; ++r) {
      int row0 = mb + r, row1 = 16 + mb + r;
      int node0 = nd0 + row0, node1 = nd0 + row1;
      float b0 = (node0 < Nn) ? h[(size_t)node0 * ND + n] : 0.f;
      float b1 = (node1 < Nn) ? h[(size_t)node1 * ND + n] : 0.f;
      hr[row0 * ND + n] = b0 + acc0[r] + bb;
      hr[row1 * ND + n] = b1 + acc1[r] + bb;
    }
  }
  __syncthreads();
  { // LayerNorm over ND=128; 8 lanes per node, shfl_xor reductions
    int e = tid >> 3, j = tid & 7;
    float s = 0.f, s2 = 0.f;
#pragma unroll
    for (int t = 0; t < 16; ++t) { float v = hr[e * ND + j + t * 8]; s += v; s2 += v * v; }
#pragma unroll
    for (int off = 4; off; off >>= 1) { s += __shfl_xor(s, off, 32); s2 += __shfl_xor(s2, off, 32); }
    float mu   = s * (1.f / ND);
    float var  = s2 * (1.f / ND) - mu * mu;
    float rstd = rsqrtf(var + 1e-5f);
    int node = nd0 + e;
    if (node < Nn) {
#pragma unroll
      for (int t = 0; t < 16; ++t) {
        int c = j + t * 8;
        float v = hr[e * ND + c];
        h_out[(size_t)node * ND + c] = (v - mu) * rstd * gamma[c] + beta[c];
      }
    }
  }
}

// ---------------------------------------------------------------------------
// Helpers: weight conversion, zeroing, coordinate update
// ---------------------------------------------------------------------------
__global__ void cvt_bf16(const float* __restrict__ src, unsigned short* __restrict__ dst,
                         int rows_src, int rows_dst, int cols) {
  int i = blockIdx.x * blockDim.x + threadIdx.x;
  if (i >= rows_dst * cols) return;
  int r = i / cols;
  dst[i] = (r < rows_src) ? f2bf(src[i]) : (unsigned short)0;
}

__global__ void zero_f32(float* __restrict__ p, long long n) {
  long long i = (long long)blockIdx.x * blockDim.x + threadIdx.x;
  if (i < n) p[i] = 0.f;
}

__global__ void egnn_coord_kernel(const float* __restrict__ x, const float* __restrict__ cagg,
                                  float* __restrict__ x_out, int n3, float invdeg) {
  int i = blockIdx.x * blockDim.x + threadIdx.x;
  if (i < n3) x_out[i] = x[i] + cagg[i] * invdeg;
}

// ---------------------------------------------------------------------------
extern "C" void kernel_launch(void* const* d_in, const int* in_sizes, int n_in,
                              void* d_out, int out_size, void* d_ws, size_t ws_size,
                              hipStream_t stream) {
  (void)n_in; (void)out_size; (void)ws_size;
  const float* h    = (const float*)d_in[0];
  const float* x    = (const float*)d_in[1];
  const float* ea   = (const float*)d_in[2];
  const int*   eidx = (const int*)d_in[3];
  const float* We1  = (const float*)d_in[4];
  const float* be1  = (const float*)d_in[5];
  const float* We2  = (const float*)d_in[6];
  const float* be2  = (const float*)d_in[7];
  const float* Wa   = (const float*)d_in[8];
  const float* ba   = (const float*)d_in[9];
  const float* Wc1  = (const float*)d_in[10];
  const float* bc1  = (const float*)d_in[11];
  const float* Wc2  = (const float*)d_in[12];
  const float* Wn1  = (const float*)d_in[13];
  const float* bn1  = (const float*)d_in[14];
  const float* Wn2  = (const float*)d_in[15];
  const float* bn2  = (const float*)d_in[16];
  const float* gamma= (const float*)d_in[17];
  const float* beta = (const float*)d_in[18];

  const int Nn = in_sizes[0] / ND;   // 20000
  const int Ne = in_sizes[3] / 2;    // 320000

  char* ws = (char*)d_ws;
  size_t off = 0;
  auto carve = [&](size_t bytes) -> char* {
    char* p = ws + off;
    off = (off + bytes + 255) & ~(size_t)255;
    return p;
  };
  unsigned short* We1b = (unsigned short*)carve((size_t)KE * HD * 2);
  unsigned short* We2b = (unsigned short*)carve((size_t)HD * HD * 2);
  unsigned short* Wc1b = (unsigned short*)carve((size_t)HD * HD * 2);
  unsigned short* Wn1b = (unsigned short*)carve((size_t)KN * HD * 2);
  unsigned short* Wn2b = (unsigned short*)carve((size_t)HD * ND * 2);
  float* agg  = (float*)carve((size_t)Nn * HD * 4);
  float* cagg = (float*)carve((size_t)Nn * 3 * 4);

  auto cdiv = [](long long a, long long b) { return (int)((a + b - 1) / b); };

  cvt_bf16<<<cdiv(KE * HD, 256), 256, 0, stream>>>(We1, We1b, 2 * ND + 1 + ED, KE, HD);
  cvt_bf16<<<cdiv(HD * HD, 256), 256, 0, stream>>>(We2, We2b, HD, HD, HD);
  cvt_bf16<<<cdiv(HD * HD, 256), 256, 0, stream>>>(Wc1, Wc1b, HD, HD, HD);
  cvt_bf16<<<cdiv(KN * HD, 256), 256, 0, stream>>>(Wn1, Wn1b, KN, KN, HD);
  cvt_bf16<<<cdiv(HD * ND, 256), 256, 0, stream>>>(Wn2, Wn2b, HD, HD, ND);

  zero_f32<<<cdiv((long long)Nn * HD, 256), 256, 0, stream>>>(agg, (long long)Nn * HD);
  zero_f32<<<cdiv((long long)Nn * 3, 256), 256, 0, stream>>>(cagg, (long long)Nn * 3);

  egnn_edge_kernel<<<cdiv(Ne, EB), 256, 0, stream>>>(
      h, x, ea, eidx, We1b, be1, We2b, be2, Wa, ba, Wc1b, bc1, Wc2, agg, cagg, Ne);

  float* h_out = (float*)d_out;
  float* x_out = h_out + (size_t)Nn * ND;

  egnn_node_kernel<<<cdiv(Nn, NB), 256, 0, stream>>>(
      h, agg, Wn1b, bn1, Wn2b, bn2, gamma, beta, h_out, Nn);

  float deg = (float)Ne / (float)Nn;
  egnn_coord_kernel<<<cdiv((long long)Nn * 3, 256), 256, 0, stream>>>(
      x, cagg, x_out, Nn * 3, 1.f / (deg + 1e-8f));
}
